// GNN_Module_68195490726192
// MI455X (gfx1250) — matrix operand order repaired and checked by reading, not compiled, not run on hardware
//
#include <hip/hip_runtime.h>
#include <hip/hip_bf16.h>

// GNN (B=16,N=128,DH=128,L=4) for gfx1250: bf16 WMMA (f32 accum) everywhere,
// edge MLP fused with gather + SiLU + segment-sum so no [B,E,*] tensor ever
// hits HBM. Weights are pre-transposed to bf16 (N-major) so WMMA B-fragments
// are contiguous 32B loads.

#define Bm 16
#define Nm 128
#define DINm 16
#define DEm 16
#define DHm 128
#define DOUTm 128
#define NLAYER 4

typedef __bf16 bf16_t;
typedef __attribute__((ext_vector_type(16))) __bf16 bf16x16;
typedef __attribute__((ext_vector_type(8))) float floatx8;

__device__ __forceinline__ floatx8 wmma_bf16(bf16x16 a, bf16x16 b, floatx8 c) {
  // D = A(16x32) * B(32x16) + C, f32 accumulate
  return __builtin_amdgcn_wmma_f32_16x16x32_bf16(false, a, false, b, (short)0, c,
                                                 false, false);
}

// Fast SiLU: x * sigmoid(x) with v_rcp_f32 instead of IEEE division.
// exp and rcp are TRANS ops that co-execute with WMMA on CDNA5.
__device__ __forceinline__ float silu_f(float x) {
  return x * __builtin_amdgcn_rcpf(1.0f + __expf(-x));
}

// A fragment: 16(M) x 32(K) bf16, source row-major with `stride` elements/row.
// lanes 0-15 -> rows 0-15 holding K 0..15; lanes 16-31 -> same rows, K 16..31
// (contiguous, matching load_b_frag).
__device__ __forceinline__ bf16x16 load_a_frag(const bf16_t* base, int stride) {
  const int lane = threadIdx.x & 31;
  const int row = lane & 15;
  const int khalf = (lane >> 4) << 4;  // 0 or 16
  const bf16_t* p = base + row * stride + khalf;
  bf16x16 f;
  ((uint4*)&f)[0] = *(const uint4*)(p);
  ((uint4*)&f)[1] = *(const uint4*)(p + 8);
  return f;
}

// B fragment: 32(K) x 16(N) bf16, source stored TRANSPOSED (N-major): baseT[n][k]
// with `stride` elements/row. Contiguous k order (this code's): lanes 0-15 -> col = lane, K 0..15;
// lanes 16-31 -> col = lane-16, K 16..31.
__device__ __forceinline__ bf16x16 load_b_frag(const bf16_t* baseT, int stride) {
  const int lane = threadIdx.x & 31;
  const int col = lane & 15;
  const int k0 = (lane >> 4) << 4;  // 0 or 16
  const bf16_t* p = baseT + col * stride + k0;
  bf16x16 f;
  ((uint4*)&f)[0] = *(const uint4*)(p);
  ((uint4*)&f)[1] = *(const uint4*)(p + 8);
  return f;
}

// ---------------------------------------------------------------------------
// Weight prep: transpose f32 [K,128] -> bf16 [128,Kpad] (zero pad K..Kpad).
// ---------------------------------------------------------------------------
__global__ void __launch_bounds__(256) transpose_w(const float* __restrict__ src,
                                                   bf16_t* __restrict__ dst,
                                                   int K, int Kpad) {
  int idx = blockIdx.x * 256 + threadIdx.x;
  if (idx >= 128 * Kpad) return;
  int n = idx / Kpad, k = idx % Kpad;
  dst[idx] = (k < K) ? (bf16_t)src[k * 128 + n] : (bf16_t)0.0f;
}

// ---------------------------------------------------------------------------
// Embedding: h = silu(x @ ei_w1 + b1) @ ei_w2 + b2. 2048 rows, 128 M-tiles.
// ---------------------------------------------------------------------------
__global__ void __launch_bounds__(256) embed_kernel(
    const float* __restrict__ x, const bf16_t* __restrict__ w1T,  // [128,32]
    const float* __restrict__ b1, const bf16_t* __restrict__ w2T, // [128,128]
    const float* __restrict__ b2, float* __restrict__ h) {
  __shared__ __align__(16) bf16_t sX[16 * 32];
  __shared__ __align__(16) bf16_t sZ[16 * 128];
  const int m0 = blockIdx.x * 16;
  const int t = threadIdx.x;
  for (int i = t; i < 16 * 32; i += 256) {
    int r = i >> 5, k = i & 31;
    sX[i] = (k < DINm) ? (bf16_t)x[(m0 + r) * DINm + k] : (bf16_t)0.0f;
  }
  __syncthreads();
  const int wv = t >> 5, lane = t & 31, col = lane & 15, r0 = (lane >> 4) * 8;
  const int f = wv * 16 + col;
  {
    bf16x16 a = load_a_frag(sX, 32);
    bf16x16 b = load_b_frag(w1T + (wv * 16) * 32, 32);
    floatx8 acc = {};
    acc = wmma_bf16(a, b, acc);
    float bias = b1[f];
#pragma unroll
    for (int v = 0; v < 8; ++v) sZ[(r0 + v) * 128 + f] = (bf16_t)silu_f(acc[v] + bias);
  }
  __syncthreads();
  {
    floatx8 acc = {};
#pragma unroll
    for (int kk = 0; kk < 4; ++kk) {
      bf16x16 a = load_a_frag(sZ + kk * 32, 128);
      bf16x16 b = load_b_frag(w2T + (wv * 16) * 128 + kk * 32, 128);
      acc = wmma_bf16(a, b, acc);
    }
    float bias = b2[f];
#pragma unroll
    for (int v = 0; v < 8; ++v) h[(m0 + r0 + v) * 128 + f] = acc[v] + bias;
  }
}

// ---------------------------------------------------------------------------
// Per-layer precompute: ta = h @ Wr + e_b1 (y==0), tb = h @ Wc (y==1).
// ---------------------------------------------------------------------------
__global__ void __launch_bounds__(256) tatb_kernel(
    const float* __restrict__ h, const bf16_t* __restrict__ WrT,
    const bf16_t* __restrict__ WcT, const float* __restrict__ eb1,
    float* __restrict__ ta, float* __restrict__ tb) {
  __shared__ __align__(16) bf16_t sA[16 * 128];
  const int m0 = blockIdx.x * 16;
  const bool first = (blockIdx.y == 0);
  const bf16_t* WT = first ? WrT : WcT;
  float* outp = first ? ta : tb;
  const int t = threadIdx.x;
  for (int i = t; i < 16 * 128; i += 256) {
    int r = i >> 7, k = i & 127;
    sA[i] = (bf16_t)h[(m0 + r) * 128 + k];
  }
  __syncthreads();
  const int wv = t >> 5, lane = t & 31, col = lane & 15, r0 = (lane >> 4) * 8;
  floatx8 acc = {};
#pragma unroll
  for (int kk = 0; kk < 4; ++kk) {
    bf16x16 a = load_a_frag(sA + kk * 32, 128);
    bf16x16 b = load_b_frag(WT + (wv * 16) * 128 + kk * 32, 128);
    acc = wmma_bf16(a, b, acc);
  }
  const int f = wv * 16 + col;
  const float bias = first ? eb1[f] : 0.0f;
#pragma unroll
  for (int v = 0; v < 8; ++v) outp[(m0 + r0 + v) * 128 + f] = acc[v] + bias;
}

// ---------------------------------------------------------------------------
// Edge MLP, fused: one WG per (batch b, source node i). 256 threads = 8 waves.
//   pre1(c,f) = ea(i,c)@We + ta[b,i,f] + tb[b,c,f]   (e_b1 folded into ta)
//   z = silu(pre1)                                    [128 c x 128 f] in LDS
//   m = silu(z @ W2 + b2);  m_i[b,i,f] = sum_c m(c,f) (ds_add_f32 reduction)
// ---------------------------------------------------------------------------
__global__ void __launch_bounds__(256) edge_kernel(
    const float* __restrict__ edge_attr, const float* __restrict__ ta,
    const float* __restrict__ tb, const bf16_t* __restrict__ WeT,  // [128,32]
    const bf16_t* __restrict__ W2T,                                // [128,128]
    const float* __restrict__ b2, float* __restrict__ m_i) {
  __shared__ __align__(16) bf16_t sEA[128 * 32];
  __shared__ __align__(16) bf16_t sZ[128 * 128];
  __shared__ float sTa[128];
  __shared__ float sAcc[128];
  const int bi = blockIdx.x;  // b*128 + i
  const int b = bi >> 7;
  const int t = threadIdx.x;
  __builtin_prefetch(W2T, 0, 1);
  const float* ea = edge_attr + (size_t)bi * (Nm * DEm);
  for (int idx = t; idx < 128 * 32; idx += 256) {
    int c = idx >> 5, k = idx & 31;
    sEA[idx] = (k < DEm) ? (bf16_t)ea[c * DEm + k] : (bf16_t)0.0f;
  }
  if (t < 128) {
    sTa[t] = ta[(size_t)bi * 128 + t];
    sAcc[t] = 0.0f;
  }
  __syncthreads();
  const int wv = t >> 5, lane = t & 31, col = lane & 15, r0 = (lane >> 4) * 8;
  const float* tbB = tb + (size_t)b * (128 * 128);
  // GEMM1 (+gather terms, SiLU): wave wv owns edge rows c in [16wv, 16wv+16)
  {
    bf16x16 a = load_a_frag(sEA + (wv * 16) * 32, 32);
#pragma unroll
    for (int nt = 0; nt < 8; ++nt) {
      bf16x16 bb = load_b_frag(WeT + (nt * 16) * 32, 32);
      floatx8 acc = {};
      acc = wmma_bf16(a, bb, acc);
      int f = nt * 16 + col;
      float tav = sTa[f];
#pragma unroll
      for (int v = 0; v < 8; ++v) {
        int c = wv * 16 + r0 + v;
        sZ[c * 128 + f] = (bf16_t)silu_f(acc[v] + tav + tbB[c * 128 + f]);
      }
    }
  }
  __syncthreads();
  // GEMM2 + SiLU + column-sum over edges
  {
    bf16x16 af[4];
#pragma unroll
    for (int kk = 0; kk < 4; ++kk) af[kk] = load_a_frag(sZ + (wv * 16) * 128 + kk * 32, 128);
#pragma unroll
    for (int nt = 0; nt < 8; ++nt) {
      floatx8 acc = {};
#pragma unroll
      for (int kk = 0; kk < 4; ++kk) {
        bf16x16 bb = load_b_frag(W2T + (nt * 16) * 128 + kk * 32, 128);
        acc = wmma_bf16(af[kk], bb, acc);
      }
      int f = nt * 16 + col;
      float bias = b2[f];
      float s = 0.0f;
#pragma unroll
      for (int v = 0; v < 8; ++v) s += silu_f(acc[v] + bias);
      atomicAdd(&sAcc[f], s);  // ds_add_f32
    }
  }
  __syncthreads();
  if (t < 128) m_i[(size_t)bi * 128 + t] = sAcc[t];
}

// ---------------------------------------------------------------------------
// Node MLP with residual: h += silu([h, m_i] @ n_w1 + b1) @ n_w2 + b2.
// ---------------------------------------------------------------------------
__global__ void __launch_bounds__(256) node_kernel(
    float* __restrict__ h, const float* __restrict__ mi,
    const bf16_t* __restrict__ W1T,  // [128,256]
    const float* __restrict__ b1, const bf16_t* __restrict__ W2T,  // [128,128]
    const float* __restrict__ b2) {
  __shared__ __align__(16) bf16_t sA[16 * 256];
  __shared__ __align__(16) bf16_t sZ[16 * 128];
  const int m0 = blockIdx.x * 16;
  const int t = threadIdx.x;
  for (int i = t; i < 16 * 256; i += 256) {
    int r = i >> 8, k = i & 255;
    float v = (k < 128) ? h[(m0 + r) * 128 + k] : mi[(m0 + r) * 128 + (k - 128)];
    sA[i] = (bf16_t)v;
  }
  __syncthreads();
  const int wv = t >> 5, lane = t & 31, col = lane & 15, r0 = (lane >> 4) * 8;
  const int f = wv * 16 + col;
  {
    floatx8 acc = {};
#pragma unroll
    for (int kk = 0; kk < 8; ++kk) {
      bf16x16 a = load_a_frag(sA + kk * 32, 256);
      bf16x16 b = load_b_frag(W1T + (wv * 16) * 256 + kk * 32, 256);
      acc = wmma_bf16(a, b, acc);
    }
    float bias = b1[f];
#pragma unroll
    for (int v = 0; v < 8; ++v) sZ[(r0 + v) * 128 + f] = (bf16_t)silu_f(acc[v] + bias);
  }
  __syncthreads();
  {
    floatx8 acc = {};
#pragma unroll
    for (int kk = 0; kk < 4; ++kk) {
      bf16x16 a = load_a_frag(sZ + kk * 32, 128);
      bf16x16 b = load_b_frag(W2T + (wv * 16) * 128 + kk * 32, 128);
      acc = wmma_bf16(a, b, acc);
    }
    float bias = b2[f];
#pragma unroll
    for (int v = 0; v < 8; ++v) {
      int row = m0 + r0 + v;
      h[row * 128 + f] += acc[v] + bias;  // residual
    }
  }
}

// ---------------------------------------------------------------------------
// Readout: out = silu(h.reshape(16,16384) @ eo_w1 + b1) @ eo_w2 + b2.
// Single WG; M = 16 batches = one M-tile; K = 16384 streamed.
// ---------------------------------------------------------------------------
__global__ void __launch_bounds__(256) final_kernel(
    const float* __restrict__ h, const bf16_t* __restrict__ w1T,  // [128,16384]
    const float* __restrict__ b1, const bf16_t* __restrict__ w2T, // [128,128]
    const float* __restrict__ b2, float* __restrict__ out) {
  __shared__ __align__(16) bf16_t sZ[16 * 128];
  const int t = threadIdx.x;
  const int wv = t >> 5, lane = t & 31, col = lane & 15, r0 = (lane >> 4) * 8;
  const int row = lane & 15;            // batch
  const int khalf = (lane >> 4) * 16;
  const int n = wv * 16 + col;
  floatx8 acc = {};
  for (int kk = 0; kk < 512; ++kk) {
    int k0 = kk * 32;
    const float* pa = h + row * 16384 + k0 + khalf;
    float4 q0 = *(const float4*)(pa);
    float4 q1 = *(const float4*)(pa + 4);
    float4 q2 = *(const float4*)(pa + 8);
    float4 q3 = *(const float4*)(pa + 12);
    bf16x16 a;
    a[0] = (bf16_t)q0.x; a[1] = (bf16_t)q0.y; a[2] = (bf16_t)q0.z; a[3] = (bf16_t)q0.w;
    a[4] = (bf16_t)q1.x; a[5] = (bf16_t)q1.y; a[6] = (bf16_t)q1.z; a[7] = (bf16_t)q1.w;
    a[8]  = (bf16_t)q2.x; a[9]  = (bf16_t)q2.y; a[10] = (bf16_t)q2.z; a[11] = (bf16_t)q2.w;
    a[12] = (bf16_t)q3.x; a[13] = (bf16_t)q3.y; a[14] = (bf16_t)q3.z; a[15] = (bf16_t)q3.w;
    bf16x16 bb;
    const bf16_t* pb = w1T + n * 16384 + k0 + (lane >> 4) * 16;
    ((uint4*)&bb)[0] = *(const uint4*)(pb);
    ((uint4*)&bb)[1] = *(const uint4*)(pb + 8);
    acc = wmma_bf16(a, bb, acc);
  }
  {
    float bias = b1[n];
#pragma unroll
    for (int v = 0; v < 8; ++v) sZ[(r0 + v) * 128 + n] = (bf16_t)silu_f(acc[v] + bias);
  }
  __syncthreads();
  {
    floatx8 acc2 = {};
#pragma unroll
    for (int kk = 0; kk < 4; ++kk) {
      bf16x16 a = load_a_frag(sZ + kk * 32, 128);
      bf16x16 b = load_b_frag(w2T + (wv * 16) * 128 + kk * 32, 128);
      acc2 = wmma_bf16(a, b, acc2);
    }
    float bias = b2[n];
#pragma unroll
    for (int v = 0; v < 8; ++v) out[(r0 + v) * 128 + n] = acc2[v] + bias;
  }
}

// ---------------------------------------------------------------------------
// Host launcher. Input order = setup_inputs() insertion order, nested dicts
// flattened depth-first in insertion order:
//  0 x, 1 edge_attr, 2..5 ei_w1/ei_b1/ei_w2/ei_b2,
//  6+8l .. : e_w1,e_b1,e_w2,e_b2,n_w1,n_b1,n_w2,n_b2 (l=0..3),
//  38..41 eo_w1/eo_b1/eo_w2/eo_b2, 42 r, 43 c (r/c unused: fully-connected).
// ---------------------------------------------------------------------------
extern "C" void kernel_launch(void* const* d_in, const int* in_sizes, int n_in,
                              void* d_out, int out_size, void* d_ws, size_t ws_size,
                              hipStream_t stream) {
  (void)in_sizes; (void)n_in; (void)out_size; (void)ws_size;
  const float* x     = (const float*)d_in[0];
  const float* eattr = (const float*)d_in[1];
  const float* ei_w1 = (const float*)d_in[2];
  const float* ei_b1 = (const float*)d_in[3];
  const float* ei_w2 = (const float*)d_in[4];
  const float* ei_b2 = (const float*)d_in[5];
  const float* eo_w1 = (const float*)d_in[38];
  const float* eo_b1 = (const float*)d_in[39];
  const float* eo_w2 = (const float*)d_in[40];
  const float* eo_b2 = (const float*)d_in[41];
  float* out = (float*)d_out;

  char* ws = (char*)d_ws;
  size_t off = 0;
  auto carve = [&](size_t bytes) -> char* {
    char* p = ws + off;
    off = (off + bytes + 255) & ~(size_t)255;
    return p;
  };
  float* h   = (float*)carve(2048 * 128 * sizeof(float));
  float* ta  = (float*)carve(2048 * 128 * sizeof(float));
  float* tb  = (float*)carve(2048 * 128 * sizeof(float));
  float* mi  = (float*)carve(2048 * 128 * sizeof(float));
  bf16_t* ei_w1T = (bf16_t*)carve(128 * 32 * 2);
  bf16_t* ei_w2T = (bf16_t*)carve(128 * 128 * 2);
  bf16_t* WrT[NLAYER]; bf16_t* WcT[NLAYER]; bf16_t* WeT[NLAYER];
  bf16_t* eW2T[NLAYER]; bf16_t* nW1T[NLAYER]; bf16_t* nW2T[NLAYER];
  for (int l = 0; l < NLAYER; ++l) {
    WrT[l]  = (bf16_t*)carve(128 * 128 * 2);
    WcT[l]  = (bf16_t*)carve(128 * 128 * 2);
    WeT[l]  = (bf16_t*)carve(128 * 32 * 2);
    eW2T[l] = (bf16_t*)carve(128 * 128 * 2);
    nW1T[l] = (bf16_t*)carve(128 * 256 * 2);
    nW2T[l] = (bf16_t*)carve(128 * 128 * 2);
  }
  bf16_t* eo_w1T = (bf16_t*)carve((size_t)128 * 16384 * 2);
  bf16_t* eo_w2T = (bf16_t*)carve(128 * 128 * 2);

  auto T = [&](const float* src, bf16_t* dst, int K, int Kpad) {
    int total = 128 * Kpad;
    transpose_w<<<(total + 255) / 256, 256, 0, stream>>>(src, dst, K, Kpad);
  };
  T(ei_w1, ei_w1T, 16, 32);
  T(ei_w2, ei_w2T, 128, 128);
  for (int l = 0; l < NLAYER; ++l) {
    const float* e_w1 = (const float*)d_in[6 + 8 * l + 0];
    const float* e_w2 = (const float*)d_in[6 + 8 * l + 2];
    const float* n_w1 = (const float*)d_in[6 + 8 * l + 4];
    const float* n_w2 = (const float*)d_in[6 + 8 * l + 6];
    T(e_w1,              WrT[l], 128, 128);
    T(e_w1 + 128 * 128,  WcT[l], 128, 128);
    T(e_w1 + 256 * 128,  WeT[l], 16, 32);
    T(e_w2,              eW2T[l], 128, 128);
    T(n_w1,              nW1T[l], 256, 256);
    T(n_w2,              nW2T[l], 128, 128);
  }
  T(eo_w1, eo_w1T, 16384, 16384);
  T(eo_w2, eo_w2T, 128, 128);

  embed_kernel<<<128, 256, 0, stream>>>(x, ei_w1T, ei_b1, ei_w2T, ei_b2, h);

  for (int l = 0; l < NLAYER; ++l) {
    const float* e_b1 = (const float*)d_in[6 + 8 * l + 1];
    const float* e_b2 = (const float*)d_in[6 + 8 * l + 3];
    const float* n_b1 = (const float*)d_in[6 + 8 * l + 5];
    const float* n_b2 = (const float*)d_in[6 + 8 * l + 7];
    tatb_kernel<<<dim3(128, 2), 256, 0, stream>>>(h, WrT[l], WcT[l], e_b1, ta, tb);
    edge_kernel<<<2048, 256, 0, stream>>>(eattr, ta, tb, WeT[l], eW2T[l], e_b2, mi);
    node_kernel<<<128, 256, 0, stream>>>(h, mi, nW1T[l], n_b1, nW2T[l], n_b2);
  }

  final_kernel<<<1, 256, 0, stream>>>(h, eo_w1T, eo_b1, eo_w2T, eo_b2, out);
}
